// Hypernetwork_41893111005501
// MI455X (gfx1250) — matrix operand order repaired and checked by reading, not compiled, not run on hardware
//
#include <hip/hip_runtime.h>
#include <hip/hip_bf16.h>

// ---------------- problem constants (from reference) ----------------
#define BATCH   16384
#define INPF    4096
#define MN      100
#define KN      32
#define HIDN    32
#define OUTN    10
#define L0      64
#define L1      256
#define L2N     128
#define WOUT    1386   // KN*HIDN + HIDN + HIDN*OUTN + OUTN

#define MSPLIT  4
#define MCHUNK  (MN / MSPLIT)      // 25
#define BBLKS   (BATCH / 128)      // 128 batch blocks (8 waves x 16 rows)

typedef _Float16 v16h __attribute__((ext_vector_type(16)));
typedef _Float16 v8h  __attribute__((ext_vector_type(8)));
typedef float    v8f  __attribute__((ext_vector_type(8)));

// ---------------- workspace layout (bytes, 32B aligned) ----------------
#define OFF_H0    0          // 100*64  f32  (25600)
#define OFF_H1    25600      // 100*256 f32  (102400)
#define OFF_H2    128000     // 100*128 f32  (51200)
#define OFF_TW1   179200     // 100*1024 f16 (204800)  layout [m][h][k]
#define OFF_TB1   384000     // 100*32  f32  (12800)
#define OFF_TW2   396800     // 100*512 f16  (102400)  layout [m][o(pad16)][h]
#define OFF_TB2   499200     // 100*10  f32  (4000)
#define OFF_SB2   503200     // 16 f32       (64)
#define OFF_PART  503296     // MSPLIT*BATCH*OUTN f32 (2621440)

// ============ Phase A1: h0 = relu(mask @ Wi.T + bi)  (dedup'd gather-sum) ============
__global__ void hyper_a1(const float* __restrict__ Wi, const float* __restrict__ bi,
                         const int* __restrict__ idx, float* __restrict__ h0) {
    const int m = blockIdx.x;     // 100
    const int o = threadIdx.x;    // 64
    float s = bi[o];
    int prev = -1;
    #pragma unroll 4
    for (int k = 0; k < KN; ++k) {
        const int c = idx[m * KN + k];     // sorted -> duplicates adjacent
        if (c != prev) s += Wi[(size_t)o * INPF + c];
        prev = c;
    }
    h0[m * L0 + o] = fmaxf(s, 0.f);
}

// ============ Phase A2: h1 = relu(h0 @ W1.T + b1) ============
__global__ void hyper_a2(const float* __restrict__ h0, const float* __restrict__ W1,
                         const float* __restrict__ b1, float* __restrict__ h1) {
    const int m = blockIdx.x;     // 100
    const int j = threadIdx.x;    // 256
    const float* h = h0 + m * L0;
    const float* w = W1 + (size_t)j * L0;
    float s = b1[j];
    #pragma unroll 8
    for (int c = 0; c < L0; ++c) s += h[c] * w[c];
    h1[m * L1 + j] = fmaxf(s, 0.f);
}

// ============ Phase A3: h2 = relu(h1 @ W2.T + b2) ============
__global__ void hyper_a3(const float* __restrict__ h1, const float* __restrict__ W2,
                         const float* __restrict__ b2, float* __restrict__ h2) {
    const int m = blockIdx.x;     // 100
    const int j = threadIdx.x;    // 128
    const float* h = h1 + m * L1;
    const float* w = W2 + (size_t)j * L1;
    float s = b2[j];
    #pragma unroll 8
    for (int c = 0; c < L1; ++c) s += h[c] * w[c];
    h2[m * L2N + j] = fmaxf(s, 0.f);
}

// ============ Phase B: w = h2 @ Wo.T + bo, scatter into f16 weight tensors ============
__global__ void hyper_b(const float* __restrict__ h2, const float* __restrict__ Wo,
                        const float* __restrict__ bo,
                        _Float16* __restrict__ tW1h, float* __restrict__ tb1,
                        _Float16* __restrict__ tW2h, float* __restrict__ tb2) {
    const int m = blockIdx.x;     // 100
    __shared__ float sh[L2N];
    if (threadIdx.x < L2N) sh[threadIdx.x] = h2[m * L2N + threadIdx.x];
    __syncthreads();
    for (int j = threadIdx.x; j < WOUT; j += 256) {
        const float* w = Wo + (size_t)j * L2N;
        float s = bo[j];
        #pragma unroll 8
        for (int c = 0; c < L2N; ++c) s += sh[c] * w[c];
        if (j < KN * HIDN) {                       // tW1[m][h][k], j = h*KN + k
            tW1h[m * (KN * HIDN) + j] = (_Float16)s;
        } else if (j < KN * HIDN + HIDN) {         // tb1[m][h]
            tb1[m * HIDN + (j - KN * HIDN)] = s;
        } else if (j < KN * HIDN + HIDN + HIDN * OUTN) {   // tW2[m][o][h], t = o*HIDN + h
            const int t = j - (KN * HIDN + HIDN);
            tW2h[m * (16 * HIDN) + t] = (_Float16)s;
        } else {                                   // tb2[m][o]
            tb2[m * OUTN + (j - (KN * HIDN + HIDN + HIDN * OUTN))] = s;
        }
    }
    // zero the padded rows o = 10..15 of tW2h (deterministic every call)
    for (int t = OUTN * HIDN + threadIdx.x; t < 16 * HIDN; t += 256)
        tW2h[m * (16 * HIDN) + t] = (_Float16)0.f;
}

// ============ Phase B2: sb2[o] = sum_m tb2[m][o] ============
__global__ void hyper_b2(const float* __restrict__ tb2, float* __restrict__ sb2) {
    const int o = threadIdx.x;
    if (o < 16) {
        float s = 0.f;
        if (o < OUTN)
            for (int m = 0; m < MN; ++m) s += tb2[m * OUTN + o];
        sb2[o] = s;
    }
}

// ============ Phase C: gather + batched WMMA GEMMs, partial over an m-chunk ============
// Each wave (32 lanes) owns a 16-row batch tile; LDS tiles are per-wave private, so
// only wave-level scheduling fences are needed (CDNA5 LDS is in-order per wave).
__global__ __launch_bounds__(256) void batched_wmma(
        const float* __restrict__ data, const int* __restrict__ idx,
        const _Float16* __restrict__ tW1h, const float* __restrict__ tb1,
        const _Float16* __restrict__ tW2h,
        float* __restrict__ part) {
    __shared__ int s_idx[MN * KN];                         // 12800 B
    __shared__ __align__(32) _Float16 s_md[8][16 * KN];    // 8192 B (per-wave A tiles)
    __shared__ __align__(32) _Float16 s_th[8][16 * KN];    // 8192 B (per-wave th tiles)

    const int bblk = blockIdx.x % BBLKS;    // batch block
    const int ms   = blockIdx.x / BBLKS;    // m-chunk id

    for (int i = threadIdx.x; i < MN * KN; i += 256) s_idx[i] = idx[i];
    __syncthreads();

    const int wave = threadIdx.x >> 5;
    const int lane = threadIdx.x & 31;
    const int r    = lane & 15;    // tile row (gather) / N index (C layout)
    const int hi   = lane >> 4;    // lane half
    const int b0   = bblk * 128 + wave * 16;

    _Float16* md = s_md[wave];
    _Float16* th = s_th[wave];

    // A-operand per-lane K runs (contiguous k order used here, same as B):
    //   lanes 0-15: K = 0..15 ; lanes 16-31: K = 16..31
    const int k0 = hi ? 16 : 0;
    const int k1 = hi ? 24 : 8;

    v8f acc = {};
    const float* drow = data + (size_t)(b0 + r) * INPF;

    const int mEnd = ms * MCHUNK + MCHUNK;
    for (int m = ms * MCHUNK; m < mEnd; ++m) {
        // ---- gather md tile: lane covers row r, k in [hi*16, hi*16+16) ----
        const int* im = s_idx + m * KN;
        #pragma unroll
        for (int k = 0; k < 16; ++k) {
            const int kk = hi * 16 + k;
            md[r * KN + kk] = (_Float16)drow[im[kk]];
        }
        if (m + 1 < mEnd) {   // gfx1250 prefetch path for next-m weights
            __builtin_prefetch(tW1h + (m + 1) * (KN * HIDN), 0, 1);
            __builtin_prefetch(tW2h + (m + 1) * (16 * HIDN), 0, 1);
        }
        __builtin_amdgcn_wave_barrier();   // fence: md stores before cross-lane md loads

        // ---- A = md tile (16x32 f16) ----
        v8h alo = *(const v8h*)(md + r * KN + k0);
        v8h ahi = *(const v8h*)(md + r * KN + k1);
        v16h a;
        #pragma unroll
        for (int i = 0; i < 8; ++i) { a[i] = alo[i]; a[8 + i] = ahi[i]; }

        // ---- B1 = tW1[m].T as (K=32 x N=16) x2 : lane n reads tW1[m][n][krun] ----
        const _Float16* w1m = tW1h + m * (KN * HIDN);
        v16h b1a = *(const v16h*)(w1m + r * KN + hi * 16);          // h = 0..15
        v16h b1b = *(const v16h*)(w1m + (16 + r) * KN + hi * 16);   // h = 16..31

        v8f c0 = {}, c1 = {};
        c0 = __builtin_amdgcn_wmma_f32_16x16x32_f16(false, a, false, b1a, (short)0, c0, false, false);
        c1 = __builtin_amdgcn_wmma_f32_16x16x32_f16(false, a, false, b1b, (short)0, c1, false, false);

        // ---- bias + relu -> th tile (f16) in LDS (transpose C->A layout) ----
        const float bb0 = tb1[m * HIDN + r];
        const float bb1 = tb1[m * HIDN + 16 + r];
        const int rowbase = hi * 8;
        #pragma unroll
        for (int i = 0; i < 8; ++i) {
            float t0 = c0[i] + bb0; t0 = t0 > 0.f ? t0 : 0.f;
            float t1 = c1[i] + bb1; t1 = t1 > 0.f ? t1 : 0.f;
            th[(rowbase + i) * KN + r]      = (_Float16)t0;
            th[(rowbase + i) * KN + 16 + r] = (_Float16)t1;
        }
        __builtin_amdgcn_wave_barrier();   // fence: th stores before cross-lane th loads

        // ---- A2 = th tile, B2 = tW2[m].T (K=h 32 x N=o 16), accumulate over m ----
        v8h a2lo = *(const v8h*)(th + r * KN + k0);
        v8h a2hi = *(const v8h*)(th + r * KN + k1);
        v16h a2;
        #pragma unroll
        for (int i = 0; i < 8; ++i) { a2[i] = a2lo[i]; a2[8 + i] = a2hi[i]; }

        const _Float16* w2m = tW2h + m * (16 * HIDN);
        v16h b2 = *(const v16h*)(w2m + r * HIDN + hi * 16);

        acc = __builtin_amdgcn_wmma_f32_16x16x32_f16(false, a2, false, b2, (short)0, acc, false, false);
        __builtin_amdgcn_wave_barrier();   // fence: tile reads done before next-m overwrite
    }

    // ---- write f32 partials: part[ms][b][o] ----
    if (r < OUTN) {
        const int rowbase = hi * 8;
        float* p = part + (size_t)ms * BATCH * OUTN;
        #pragma unroll
        for (int i = 0; i < 8; ++i)
            p[(size_t)(b0 + rowbase + i) * OUTN + r] = acc[i];
    }
}

// ============ Phase D: reduce partials, add summed bias, scale by 1/MN ============
__global__ void reduce_out(const float* __restrict__ part, const float* __restrict__ sb2,
                           float* __restrict__ out) {
    const int i = blockIdx.x * 256 + threadIdx.x;   // BATCH*OUTN elements
    if (i < BATCH * OUTN) {
        float s = 0.f;
        #pragma unroll
        for (int ms = 0; ms < MSPLIT; ++ms)
            s += part[(size_t)ms * BATCH * OUTN + i];
        out[i] = (s + sb2[i % OUTN]) * (1.0f / (float)MN);
    }
}

// ---------------- launcher ----------------
extern "C" void kernel_launch(void* const* d_in, const int* in_sizes, int n_in,
                              void* d_out, int out_size, void* d_ws, size_t ws_size,
                              hipStream_t stream) {
    const float* data = (const float*)d_in[0];
    const int*   idx  = (const int*)  d_in[1];
    const float* Wi   = (const float*)d_in[2];
    const float* bi   = (const float*)d_in[3];
    const float* W1   = (const float*)d_in[4];
    const float* b1   = (const float*)d_in[5];
    const float* W2   = (const float*)d_in[6];
    const float* b2   = (const float*)d_in[7];
    const float* Wo   = (const float*)d_in[8];
    const float* bo   = (const float*)d_in[9];
    float* out = (float*)d_out;

    char* ws = (char*)d_ws;
    float*    h0   = (float*)   (ws + OFF_H0);
    float*    h1   = (float*)   (ws + OFF_H1);
    float*    h2   = (float*)   (ws + OFF_H2);
    _Float16* tW1h = (_Float16*)(ws + OFF_TW1);
    float*    tb1  = (float*)   (ws + OFF_TB1);
    _Float16* tW2h = (_Float16*)(ws + OFF_TW2);
    float*    tb2  = (float*)   (ws + OFF_TB2);
    float*    sb2  = (float*)   (ws + OFF_SB2);
    float*    part = (float*)   (ws + OFF_PART);

    hyper_a1<<<MN, L0,  0, stream>>>(Wi, bi, idx, h0);
    hyper_a2<<<MN, L1,  0, stream>>>(h0, W1, b1, h1);
    hyper_a3<<<MN, L2N, 0, stream>>>(h1, W2, b2, h2);
    hyper_b <<<MN, 256, 0, stream>>>(h2, Wo, bo, tW1h, tb1, tW2h, tb2);
    hyper_b2<<<1, 32,   0, stream>>>(tb2, sb2);
    batched_wmma<<<BBLKS * MSPLIT, 256, 0, stream>>>(data, idx, tW1h, tb1, tW2h, part);
    reduce_out<<<(BATCH * OUTN + 255) / 256, 256, 0, stream>>>(part, sb2, out);
}
